// BG_DGAT_24575802868133
// MI455X (gfx1250) — matrix operand order repaired and checked by reading, not compiled, not run on hardware
//
#include <hip/hip_runtime.h>

typedef __attribute__((ext_vector_type(16))) __bf16 v16bf;
typedef __attribute__((ext_vector_type(8)))  float  v8f;
typedef __attribute__((ext_vector_type(4)))  unsigned int u32x4;
typedef __attribute__((ext_vector_type(4)))  float  f32x4;

#define B_      16
#define CH_     62
#define T_      2048
#define NT_     64
#define CONCAT_ 1600
#define FEAT_   51200
#define EPS_    1e-5f

// ---------------- stage 0: pack x into bf16 pair-dwords ----------------
__global__ void k_xpair(const float* __restrict__ x, unsigned int* __restrict__ xp){
  int idx = blockIdx.x*256 + threadIdx.x;
  if(idx >= B_*CH_*T_) return;
  int t = idx % T_;
  union { __bf16 h[2]; unsigned int u; } p;
  p.h[0] = (__bf16)x[idx];
  p.h[1] = (t+1 < T_) ? (__bf16)x[idx+1] : (__bf16)0.f;
  xp[idx] = p.u;
}

__global__ void k_wbf(const float* __restrict__ w1, const float* __restrict__ w2,
                      const float* __restrict__ w3, __bf16* __restrict__ wbf){
  int idx = blockIdx.x*256 + threadIdx.x;
  if(idx >= 64*(128+256+512)) return;
  float v;
  if(idx < 8192)        v = w1[idx];
  else if(idx < 24576)  v = w2[idx-8192];
  else                  v = w3[idx-24576];
  wbf[idx] = (__bf16)v;
}

// ---------------- stage 1: WMMA FIR conv + square + pool16/4 + log ----------------
__global__ __launch_bounds__(256) void k_conv(
    const unsigned int* __restrict__ xp, const __bf16* __restrict__ wbf,
    const float* __restrict__ bias, __bf16* __restrict__ concat,
    int K, int P, int off)
{
  __shared__ unsigned int sD[672];       // pair-dwords: 160 t positions + K taps
  __shared__ float sC[64*164];           // squared conv values, 64 rows x 160 t
  const int tid  = threadIdx.x;
  const int tile = blockIdx.x, bc = blockIdx.y;
  const int tbase = tile*128;            // conv-t base (pooled base = tile*32)
  const int lenD  = K + 160;
  const unsigned int* row = xp + (size_t)bc*T_;
  for(int i = tid; i < lenD; i += 256){
    int t = tbase + i;
    sD[i] = (t < T_) ? row[t] : 0u;
  }
  __syncthreads();

  const int w  = tid >> 5, l = tid & 31;
  const int ln = l & 15,  hl = l >> 4;
  const int m  = w & 3;                  // M-tile (16 NT rows)
  const int ntb = (w >> 2)*5;            // N-half: tiles [ntb, ntb+5)

  v8f acc[5];
  for(int i=0;i<5;i++) for(int j=0;j<8;j++) acc[i][j] = 0.f;

  const int rowA = m*16 + ln;
  const __bf16* wr = wbf + (size_t)rowA*K;

  for(int k0 = 0; k0 < K; k0 += 32){
    union { v16bf v; u32x4 q[2]; } A;
    A.q[0] = *(const u32x4*)(wr + k0 + hl*16);       // K = hl*16 .. +7
    A.q[1] = *(const u32x4*)(wr + k0 + hl*16 + 8);   // K = hl*16+8 .. +7
    __builtin_prefetch(wr + k0 + 32, 0, 1);
    #pragma unroll
    for(int nt = 0; nt < 5; nt++){
      const int n    = (ntb + nt)*16 + ln;
      const int base = n + k0 + hl*16;
      union { v16bf v; unsigned int u[8]; } Bf;      // B[k,n] = x[n+k] (Hankel)
      #pragma unroll
      for(int j=0;j<8;j++) Bf.u[j] = sD[base + 2*j];
      acc[nt] = __builtin_amdgcn_wmma_f32_16x16x32_bf16(
          false, A.v, false, Bf.v, (short)0, acc[nt], false, false);
    }
  }

  float bv[8];
  #pragma unroll
  for(int r=0;r<8;r++) bv[r] = bias[m*16 + r + 8*hl];
  #pragma unroll
  for(int nt=0;nt<5;nt++){
    const int tl = (ntb+nt)*16 + ln;
    #pragma unroll
    for(int r=0;r<8;r++){
      const int rr = m*16 + r + 8*hl;                // C layout: M = vgpr + 8*halflane
      float v = acc[nt][r] + bv[r];
      sC[rr*164 + tl] = v*v;
    }
  }
  __syncthreads();

  for(int it=0; it<8; it++){
    int idx = tid + it*256;                          // 2048 pooled outputs / block
    int r = idx >> 5, j = idx & 31;
    int jg = tile*32 + j;
    if(jg < P){
      float s = 0.f;
      #pragma unroll
      for(int q=0;q<16;q++) s += sC[r*164 + 4*j + q];
      // concat layout: [b][ch][t][c], channel-innermost for the oxo GEMM
      concat[((size_t)bc*CONCAT_ + off + jg)*64 + r] = (__bf16)logf(s*(1.f/16.f));
    }
  }
}

// ---------------- stage 2: pointwise t4 branch ----------------
__global__ void k_t4(const float* __restrict__ sg, const float* __restrict__ w4,
                     const float* __restrict__ b4, __bf16* __restrict__ concat){
  int idx = blockIdx.x*256 + threadIdx.x;
  if(idx >= B_*CH_*297*64) return;
  int nt = idx & 63;
  int r  = idx >> 6;
  int t  = r % 297;
  int bc = r / 297;
  float v = sg[(size_t)bc*297 + t]*w4[nt] + b4[nt];
  concat[((size_t)bc*CONCAT_ + 1303 + t)*64 + nt] = (__bf16)v;
}

// ---------------- stage 3: bn stats over concat (per NT channel) ----------------
__global__ void k_stats_concat(const __bf16* __restrict__ concat, float* mean, float* var){
  __shared__ float rs[256], rq[256];
  const int c = blockIdx.x;
  const size_t N = (size_t)B_*CH_*CONCAT_;
  float s=0.f, q=0.f;
  for(size_t i = threadIdx.x; i < N; i += 256){
    float v = (float)concat[i*64 + c];
    s += v; q += v*v;
  }
  rs[threadIdx.x]=s; rq[threadIdx.x]=q;
  __syncthreads();
  for(int o=128;o>0;o>>=1){
    if(threadIdx.x<o){ rs[threadIdx.x]+=rs[threadIdx.x+o]; rq[threadIdx.x]+=rq[threadIdx.x+o]; }
    __syncthreads();
  }
  if(threadIdx.x==0){
    float mu = rs[0]/(float)N;
    mean[c]=mu; var[c]=rq[0]/(float)N - mu*mu;
  }
}

// ---------------- stage 4: fold bn into 1x1 weights ----------------
__global__ void k_fold(const float* __restrict__ woxo, const float* __restrict__ boxo,
                       const float* __restrict__ g, const float* __restrict__ b,
                       const float* __restrict__ mean, const float* __restrict__ var,
                       __bf16* __restrict__ Wp, float* __restrict__ bp){
  int o = threadIdx.x;
  if(o >= 64) return;
  float bb = boxo[o];
  for(int c=0;c<64;c++){
    float sc = g[c]*rsqrtf(var[c]+EPS_);
    float tc = b[c] - mean[c]*sc;
    float wv = woxo[o*64+c];
    Wp[o*64+c] = (__bf16)(wv*sc);
    bb += wv*tc;
  }
  bp[o] = bb;
}

// ---------------- stage 5: WMMA 1x1 conv + leaky + pool2 ----------------
__global__ __launch_bounds__(256) void k_oxo(
    const __bf16* __restrict__ concat, const __bf16* __restrict__ Wp,
    const float* __restrict__ bp, __bf16* __restrict__ z)
{
  __shared__ float sY[64*164];
  const int tid = threadIdx.x;
  const int tile = blockIdx.x, bc = blockIdx.y;
  const int b = bc / CH_, ch = bc % CH_;
  const int w = tid>>5, l = tid&31, ln = l&15, hl = l>>4;
  const int m = w&3, ntb = (w>>2)*5;

  v8f acc[5];
  for(int i=0;i<5;i++) for(int j=0;j<8;j++) acc[i][j]=0.f;
  const int rowA = m*16 + ln;
  const __bf16* wr = Wp + rowA*64;
  #pragma unroll
  for(int k0=0;k0<64;k0+=32){
    union { v16bf v; u32x4 q[2]; } A;
    A.q[0] = *(const u32x4*)(wr + k0 + hl*16);
    A.q[1] = *(const u32x4*)(wr + k0 + hl*16 + 8);
    #pragma unroll
    for(int nt=0;nt<5;nt++){
      const int t = tile*160 + (ntb+nt)*16 + ln;
      const u32x4* bpv = (const u32x4*)(concat + ((size_t)bc*CONCAT_ + t)*64 + k0 + hl*16);
      union { v16bf v; u32x4 q[2]; } Bf;
      Bf.q[0]=bpv[0]; Bf.q[1]=bpv[1];
      acc[nt] = __builtin_amdgcn_wmma_f32_16x16x32_bf16(
          false, A.v, false, Bf.v, (short)0, acc[nt], false, false);
    }
  }
  float bv[8];
  #pragma unroll
  for(int r=0;r<8;r++) bv[r] = bp[m*16 + r + 8*hl];
  #pragma unroll
  for(int nt=0;nt<5;nt++){
    int tl = (ntb+nt)*16 + ln;
    #pragma unroll
    for(int r=0;r<8;r++){
      int rr = m*16 + r + 8*hl;
      float y = acc[nt][r] + bv[r];
      sY[rr*164 + tl] = y > 0.f ? y : 0.01f*y;
    }
  }
  __syncthreads();
  for(int it=0; it<20; it++){
    int idx = tid + it*256;                  // 64 x 80 pooled outputs
    int r = idx/80, tp_l = idx%80;
    float zz = 0.5f*(sY[r*164 + 2*tp_l] + sY[r*164 + 2*tp_l + 1]);
    int tp = tile*80 + tp_l;
    z[(((size_t)(b*64 + r))*800 + tp)*CH_ + ch] = (__bf16)zz;   // [b][nt][tp][ch]
  }
}

// ---------------- stage 6: bn stats over z (per NT channel) ----------------
__global__ void k_stats_z(const __bf16* __restrict__ z, float* mean, float* var){
  __shared__ float rs[256], rq[256];
  const int c = blockIdx.x;
  const int PER = 800*CH_;
  const size_t N = (size_t)B_*PER;
  float s=0.f,q=0.f;
  for(size_t i=threadIdx.x; i<N; i+=256){
    size_t bi = i / PER, rem = i % PER;
    float v = (float)z[((size_t)bi*64 + c)*PER + rem];
    s+=v; q+=v*v;
  }
  rs[threadIdx.x]=s; rq[threadIdx.x]=q;
  __syncthreads();
  for(int o=128;o>0;o>>=1){
    if(threadIdx.x<o){ rs[threadIdx.x]+=rs[threadIdx.x+o]; rq[threadIdx.x]+=rq[threadIdx.x+o]; }
    __syncthreads();
  }
  if(threadIdx.x==0){
    float mu = rs[0]/(float)N;
    mean[c]=mu; var[c]=rq[0]/(float)N - mu*mu;
  }
}

// ---------------- stage 7: bn2 + local filter + relu + segment mean ----------------
__global__ void k_agg(const __bf16* __restrict__ z, const float* __restrict__ lfw,
                      const float* __restrict__ lfb, const float* __restrict__ g2,
                      const float* __restrict__ b2, const float* __restrict__ mean2,
                      const float* __restrict__ var2, float* __restrict__ agg){
  int f = blockIdx.x*256 + threadIdx.x;    // exact: 200*256 == 51200
  int b = blockIdx.y;
  int nt = f/800, tp = f - nt*800;
  float so = g2[nt]*rsqrtf(var2[nt]+EPS_);
  float to = b2[nt] - mean2[nt]*so;
  const __bf16* zr = z + ((size_t)(b*64+nt)*800 + tp)*CH_;
  float a7[7]; for(int i=0;i<7;i++) a7[i]=0.f;
  #pragma unroll
  for(int ch=0; ch<CH_; ch++){
    float zv = (float)zr[ch]*so + to;
    float u = zv * lfw[(size_t)ch*FEAT_ + f] - lfb[ch];
    a7[ch/9] += (u > 0.f ? u : 0.f);
  }
  #pragma unroll
  for(int a=0;a<7;a++){
    float inv = (a<6) ? (1.f/9.f) : (1.f/8.f);
    agg[((size_t)b*7 + a)*FEAT_ + f] = a7[a]*inv;
  }
}

// ---------------- stage 8: bn1 stats over agg (per area) ----------------
__global__ void k_stats_agg(const float* __restrict__ agg, float* mean, float* var){
  __shared__ float rs[256], rq[256];
  const int a = blockIdx.x;
  const size_t N = (size_t)B_*FEAT_;
  float s=0.f,q=0.f;
  for(size_t i=threadIdx.x; i<N; i+=256){
    size_t b = i / FEAT_, f = i % FEAT_;
    float v = agg[((size_t)b*7 + a)*FEAT_ + f];
    s+=v; q+=v*v;
  }
  rs[threadIdx.x]=s; rq[threadIdx.x]=q;
  __syncthreads();
  for(int o=128;o>0;o>>=1){
    if(threadIdx.x<o){ rs[threadIdx.x]+=rs[threadIdx.x+o]; rq[threadIdx.x]+=rq[threadIdx.x+o]; }
    __syncthreads();
  }
  if(threadIdx.x==0){
    float mu = rs[0]/(float)N;
    mean[a]=mu; var[a]=rq[0]/(float)N - mu*mu;
  }
}

// ---------------- stage 9: Gram matrix s[b,i,j] ----------------
__global__ void k_gram(const float* __restrict__ agg, float* __restrict__ sgm){
  __shared__ float rs[256];
  int bi = blockIdx.x;
  int b = bi/49, r = bi%49, i = r/7, j = r%7;
  const float* A  = agg + ((size_t)b*7+i)*FEAT_;
  const float* Bm = agg + ((size_t)b*7+j)*FEAT_;
  float s=0.f;
  for(int f=threadIdx.x; f<FEAT_; f+=256) s += A[f]*Bm[f];
  rs[threadIdx.x]=s;
  __syncthreads();
  for(int o=128;o>0;o>>=1){
    if(threadIdx.x<o) rs[threadIdx.x]+=rs[threadIdx.x+o];
    __syncthreads();
  }
  if(threadIdx.x==0) sgm[bi]=rs[0];
}

// ---------------- stage 10: P = agg @ gat_W (bn folded later) ----------------
__global__ __launch_bounds__(256) void k_proj(const float* __restrict__ agg,
                                              const float* __restrict__ W,
                                              float* __restrict__ P){
  __shared__ float red[256*32];
  int ba = blockIdx.x;
  const float* A = agg + (size_t)ba*FEAT_;
  float acc[32]; for(int k=0;k<32;k++) acc[k]=0.f;
  for(int f=threadIdx.x; f<FEAT_; f+=256){
    float av = A[f];
    const f32x4* wrow = (const f32x4*)(W + (size_t)f*32);
    #pragma unroll
    for(int q=0;q<8;q++){
      f32x4 wv = wrow[q];
      #pragma unroll
      for(int j=0;j<4;j++) acc[4*q+j] += av*wv[j];
    }
  }
  for(int k=0;k<32;k++) red[threadIdx.x*32+k]=acc[k];
  __syncthreads();
  for(int o=128;o>0;o>>=1){
    if(threadIdx.x<o)
      for(int k=0;k<32;k++) red[threadIdx.x*32+k]+=red[(threadIdx.x+o)*32+k];
    __syncthreads();
  }
  if(threadIdx.x<32) P[(size_t)ba*32+threadIdx.x]=red[threadIdx.x];
}

__global__ void k_colsum(const float* __restrict__ W, float* __restrict__ cs){
  __shared__ float rs[256];
  int o = blockIdx.x;
  float s=0.f;
  for(int f=threadIdx.x; f<FEAT_; f+=256) s += W[(size_t)f*32+o];
  rs[threadIdx.x]=s;
  __syncthreads();
  for(int off=128;off>0;off>>=1){
    if(threadIdx.x<off) rs[threadIdx.x]+=rs[threadIdx.x+off];
    __syncthreads();
  }
  if(threadIdx.x==0) cs[o]=rs[0];
}

// ---------------- stage 11: GAT tail + final bn + fc ----------------
__global__ __launch_bounds__(256) void k_tail(
    const float* __restrict__ P, const float* __restrict__ cs,
    const float* __restrict__ aggm, const float* __restrict__ aggv,
    const float* __restrict__ g1, const float* __restrict__ b1,
    const float* __restrict__ sgm, const float* __restrict__ gadj,
    const float* __restrict__ ga, const float* __restrict__ g2,
    const float* __restrict__ b2, const float* __restrict__ fcW,
    const float* __restrict__ fcb, float* __restrict__ out)
{
  __shared__ float Wh[3584], O[3584];
  __shared__ float e1[112], e2[112], Dv[112];
  __shared__ float Am[784], At[784];
  __shared__ float bm[7], bvv[7];
  int tid = threadIdx.x;
  for(int idx=tid; idx<3584; idx+=256){
    int r = idx % 224;
    int a = r / 32, o = r % 32;
    float sa = g1[a]*rsqrtf(aggv[a]+EPS_);
    float ta = b1[a] - aggm[a]*sa;
    Wh[idx] = sa*P[idx] + ta*cs[o];
  }
  for(int idx=tid; idx<784; idx+=256){
    int r = idx%49, i=r/7, j=r%7;
    float g  = gadj[i*7+j] + gadj[j*7+i];
    float sv = sgm[idx] + (i==j ? 1.f : 0.f);
    float av = sv*g;
    Am[idx] = av>0.f ? av : 0.f;
  }
  __syncthreads();
  for(int idx=tid; idx<112; idx+=256){
    float s1=0.f, s2=0.f;
    for(int o=0;o<32;o++){
      float wv = Wh[idx*32+o];
      s1 += wv*ga[o];
      s2 += wv*ga[32+o];
    }
    e1[idx]=s1; e2[idx]=s2;
    int b = idx/7, i = idx%7;
    float rsum=0.f;
    for(int j=0;j<7;j++) rsum += Am[b*49+i*7+j];
    if(rsum==0.f) rsum=1.f;
    Dv[idx] = rsqrtf(rsum);
  }
  __syncthreads();
  for(int idx=tid; idx<112; idx+=256){
    int b = idx/7, i = idx%7;
    float lg[7], mx = -1e30f;
    for(int j=0;j<7;j++){
      float adjv = Dv[b*7+i]*Am[b*49+i*7+j]*Dv[b*7+j];
      float ev = e1[b*7+i] + e2[b*7+j];
      ev = ev>0.f ? ev : 0.2f*ev;
      float v = adjv>0.f ? ev : -9e15f;
      lg[j]=v; if(v>mx) mx=v;
    }
    float ssum=0.f;
    for(int j=0;j<7;j++){ lg[j]=expf(lg[j]-mx); ssum+=lg[j]; }
    float inv = 1.f/ssum;
    for(int j=0;j<7;j++) At[b*49+i*7+j] = lg[j]*inv;
  }
  __syncthreads();
  for(int idx=tid; idx<3584; idx+=256){
    int b = idx/224, r = idx%224, i=r/32, o=r%32;
    float s=0.f;
    for(int j=0;j<7;j++) s += At[b*49+i*7+j]*Wh[(b*7+j)*32+o];
    O[idx] = s>0.f ? s : expf(s)-1.f;   // elu
  }
  __syncthreads();
  if(tid<7){
    float s=0.f,q=0.f;
    for(int b=0;b<16;b++)
      for(int o=0;o<32;o++){ float v = O[b*224 + tid*32 + o]; s+=v; q+=v*v; }
    float mu = s/512.f;
    bm[tid]=mu; bvv[tid]=q/512.f - mu*mu;
  }
  __syncthreads();
  if(tid<64){
    int b = tid>>2, k = tid&3;
    float s = fcb[k];
    for(int a=0;a<7;a++){
      float sa = g2[a]*rsqrtf(bvv[a]+EPS_);
      for(int o=0;o<32;o++){
        float nv = (O[b*224+a*32+o]-bm[a])*sa + b2[a];
        s += nv*fcW[k*224 + a*32 + o];
      }
    }
    out[b*4+k] = s;
  }
}

extern "C" void kernel_launch(void* const* d_in, const int* in_sizes, int n_in,
                              void* d_out, int out_size, void* d_ws, size_t ws_size,
                              hipStream_t stream){
  const float* x    = (const float*)d_in[0];
  const float* sgx  = (const float*)d_in[1];
  const float* w1   = (const float*)d_in[2];
  const float* b1c  = (const float*)d_in[3];
  const float* w2   = (const float*)d_in[4];
  const float* b2c  = (const float*)d_in[5];
  const float* w3   = (const float*)d_in[6];
  const float* b3c  = (const float*)d_in[7];
  const float* w4   = (const float*)d_in[8];
  const float* b4   = (const float*)d_in[9];
  const float* woxo = (const float*)d_in[10];
  const float* boxo = (const float*)d_in[11];
  const float* bntg = (const float*)d_in[12];
  const float* bntb = (const float*)d_in[13];
  const float* bnt2g= (const float*)d_in[14];
  const float* bnt2b= (const float*)d_in[15];
  const float* lfw  = (const float*)d_in[16];
  const float* lfb  = (const float*)d_in[17];
  const float* gadj = (const float*)d_in[18];
  const float* bn1g = (const float*)d_in[19];
  const float* bn1b = (const float*)d_in[20];
  const float* gatW = (const float*)d_in[21];
  const float* gata = (const float*)d_in[22];
  const float* bn2g = (const float*)d_in[23];
  const float* bn2b = (const float*)d_in[24];
  const float* fcW  = (const float*)d_in[25];
  const float* fcb  = (const float*)d_in[26];
  float* out = (float*)d_out;

  char* ws = (char*)d_ws;
  size_t off = 0;
  auto take = [&](size_t bytes){
    size_t o = off; off = (off + bytes + 255) & ~(size_t)255; return o;
  };
  unsigned int* xp  = (unsigned int*)(ws + take((size_t)B_*CH_*T_*4));
  __bf16* wbf       = (__bf16*)(ws + take((size_t)57344*2));
  __bf16* concat    = (__bf16*)(ws + take((size_t)B_*CH_*CONCAT_*64*2));
  float* mean1      = (float*)(ws + take(64*4));
  float* var1       = (float*)(ws + take(64*4));
  __bf16* Wp        = (__bf16*)(ws + take(64*64*2));
  float* bp         = (float*)(ws + take(64*4));
  __bf16* z         = (__bf16*)(ws + take((size_t)B_*64*800*CH_*2));
  float* mean2      = (float*)(ws + take(64*4));
  float* var2       = (float*)(ws + take(64*4));
  float* agg        = (float*)(ws + take((size_t)B_*7*FEAT_*4));
  float* aggm       = (float*)(ws + take(16*4));
  float* aggv       = (float*)(ws + take(16*4));
  float* sgm        = (float*)(ws + take(784*4));
  float* Pm         = (float*)(ws + take(3584*4));
  float* cs         = (float*)(ws + take(32*4));

  k_xpair<<<(B_*CH_*T_+255)/256, 256, 0, stream>>>(x, xp);
  k_wbf<<<(57344+255)/256, 256, 0, stream>>>(w1, w2, w3, wbf);
  k_conv<<<dim3(15, B_*CH_), 256, 0, stream>>>(xp, wbf,         b1c, concat, 128, 477, 0);
  k_conv<<<dim3(14, B_*CH_), 256, 0, stream>>>(xp, wbf + 8192,  b2c, concat, 256, 445, 477);
  k_conv<<<dim3(12, B_*CH_), 256, 0, stream>>>(xp, wbf + 24576, b3c, concat, 512, 381, 922);
  k_t4<<<(B_*CH_*297*64+255)/256, 256, 0, stream>>>(sgx, w4, b4, concat);
  k_stats_concat<<<64, 256, 0, stream>>>(concat, mean1, var1);
  k_fold<<<1, 64, 0, stream>>>(woxo, boxo, bntg, bntb, mean1, var1, Wp, bp);
  k_oxo<<<dim3(10, B_*CH_), 256, 0, stream>>>(concat, Wp, bp, z);
  k_stats_z<<<64, 256, 0, stream>>>(z, mean2, var2);
  k_agg<<<dim3(200, B_), 256, 0, stream>>>(z, lfw, lfb, bnt2g, bnt2b, mean2, var2, agg);
  k_stats_agg<<<7, 256, 0, stream>>>(agg, aggm, aggv);
  k_gram<<<784, 256, 0, stream>>>(agg, sgm);
  k_proj<<<112, 256, 0, stream>>>(agg, gatW, Pm);
  k_colsum<<<32, 256, 0, stream>>>(gatW, cs);
  k_tail<<<1, 256, 0, stream>>>(Pm, cs, aggm, aggv, bn1g, bn1b, sgm, gadj, gata,
                                bn2g, bn2b, fcW, fcb, out);
}
